// InferDepthNet0_51599737094666
// MI455X (gfx1250) — matrix operand order repaired and checked by reading, not compiled, not run on hardware
//
#include <hip/hip_runtime.h>
#include <hip/hip_bf16.h>
#include <math.h>

typedef _Float16 h16;
typedef __attribute__((ext_vector_type(16))) _Float16 v16h;
typedef __attribute__((ext_vector_type(8)))  _Float16 v8h;
typedef __attribute__((ext_vector_type(8)))  float    v8f;

#define CPAD 32

static constexpr int Cn = 32, Hn = 128, Wn = 160, Dn = 32, BC = 8;
static constexpr int HWn = Hn * Wn;                 // 20480
static constexpr int H2 = Hn / 2, W2 = Wn / 2, HW2 = H2 * W2;   // 5120
static constexpr int H4 = Hn / 4, W4 = Wn / 4, HW4 = H4 * W4;   // 1280
static constexpr int H8 = Hn / 8, W8 = Wn / 8, HW8 = H8 * W8;   // 320
static constexpr int OHd = 2 * Hn, OWd = 2 * Wn, OHW = OHd * OWd; // 81920
static constexpr int WSLOT = 9 * 2 * 32 * 16;       // halves per packed weight slot
static constexpr int PADPIX = (Hn + 2) * (Wn + 2);  // largest padded pixel count

// Output layout (floats, concatenated in return order)
static constexpr size_t O_DEPTH = 0;
static constexpr size_t O_PMAP  = OHW;
static constexpr size_t O_PCA   = 2 * (size_t)OHW;
static constexpr size_t O_PRES  = O_PCA + 66 * (size_t)HWn;

// ---------------------------------------------------------------------------
// Small utility kernels
// ---------------------------------------------------------------------------
__global__ void k_zero(float* __restrict__ p, float v, int n) {
  int i = blockIdx.x * blockDim.x + threadIdx.x;
  if (i < n) p[i] = v;
}

// Invert ref projection, compute rot(3x3)+trans(3) for views 1,2 -> rt[2][12]
__global__ void k_proj_setup(const float* __restrict__ pm, float* __restrict__ rt) {
  if (blockIdx.x != 0 || threadIdx.x != 0) return;
  float a[4][8];
  for (int r = 0; r < 4; ++r)
    for (int c = 0; c < 4; ++c) { a[r][c] = pm[r * 4 + c]; a[r][c + 4] = (r == c) ? 1.f : 0.f; }
  for (int col = 0; col < 4; ++col) {
    int piv = col;
    for (int r = col + 1; r < 4; ++r) if (fabsf(a[r][col]) > fabsf(a[piv][col])) piv = r;
    if (piv != col) for (int c = 0; c < 8; ++c) { float t = a[col][c]; a[col][c] = a[piv][c]; a[piv][c] = t; }
    float inv = 1.f / a[col][col];
    for (int c = 0; c < 8; ++c) a[col][c] *= inv;
    for (int r = 0; r < 4; ++r) if (r != col) {
      float f = a[r][col];
      for (int c = 0; c < 8; ++c) a[r][c] -= f * a[col][c];
    }
  }
  for (int v = 1; v < 3; ++v) {
    const float* P = pm + v * 16;
    float pr[3][4];
    for (int r = 0; r < 3; ++r)
      for (int c = 0; c < 4; ++c) {
        float s = 0.f;
        for (int k = 0; k < 4; ++k) s += P[r * 4 + k] * a[k][c + 4];
        pr[r][c] = s;
      }
    float* o = rt + (v - 1) * 12;
    for (int r = 0; r < 3; ++r) for (int c = 0; c < 3; ++c) o[r * 3 + c] = pr[r][c];
    o[9] = pr[0][3]; o[10] = pr[1][3]; o[11] = pr[2][3];
  }
}

__device__ inline float bisample(const float* __restrict__ img, int H, int W, float px, float py) {
  float x0f = floorf(px), y0f = floorf(py);
  float wx = px - x0f, wy = py - y0f;
  int x0 = (int)x0f, y0 = (int)y0f;
  float acc = 0.f;
  for (int dy = 0; dy < 2; ++dy)
    for (int dx = 0; dx < 2; ++dx) {
      int ix = x0 + dx, iy = y0 + dy;
      float wgt = (dx ? wx : 1.f - wx) * (dy ? wy : 1.f - wy);
      if (ix >= 0 && ix < W && iy >= 0 && iy < H)
        acc += img[(size_t)iy * W + ix] * wgt;
    }
  return acc;
}

// Fused homography-warp + groupwise-correlation cost: cost[d][h][w]
__global__ void k_warp_cost(const float* __restrict__ ref, const float* __restrict__ src,
                            const float* __restrict__ dvals, const float* __restrict__ rt,
                            float* __restrict__ cost, int C, int D, int H, int W) {
  int HW = H * W;
  int idx = blockIdx.x * blockDim.x + threadIdx.x;
  if (idx >= D * HW) return;
  int hw = idx % HW;
  int w = hw % W, h = hw / W;
  float dep = dvals[idx];
  float rx = rt[0] * w + rt[1] * h + rt[2];
  float ry = rt[3] * w + rt[4] * h + rt[5];
  float rz = rt[6] * w + rt[7] * h + rt[8];
  float pz = rz * dep + rt[11];
  float px = (rx * dep + rt[9]) / pz;
  float py = (ry * dep + rt[10]) / pz;
  float acc = 0.f;
  for (int c = 0; c < C; ++c)
    acc += ref[(size_t)c * HW + hw] * bisample(src + (size_t)c * HW, H, W, px, py);
  cost[idx] = acc / (float)C;
}

// Pack CHW fp32 (optionally concat of two tensors) into padded HWC f16:
// dst has (H+2)x(W+2) pixels x CPAD channels, zero halo ring, zero channel pad.
__global__ void k_pack_hwc(const float* __restrict__ a, int Ca,
                           const float* __restrict__ b, int Cb,
                           h16* __restrict__ dst, int H, int W) {
  int Wp = W + 2;
  int total = (H + 2) * Wp * CPAD;
  int idx = blockIdx.x * blockDim.x + threadIdx.x;
  if (idx >= total) return;
  int c = idx & (CPAD - 1);
  int pp = idx >> 5;
  int pw = pp % Wp, ph = pp / Wp;
  float v = 0.f;
  if (ph >= 1 && ph <= H && pw >= 1 && pw <= W) {
    int p = (ph - 1) * W + (pw - 1);
    if (c < Ca) v = a[(size_t)c * H * W + p];
    else if (c < Ca + Cb) v = b[(size_t)(c - Ca) * H * W + p];
  }
  dst[idx] = (h16)v;
}

// Pack OIHW fp32 weights into WMMA B-fragment order: [tap][ntile][lane][i]
__global__ void k_pack_wfrag(const float* __restrict__ w, h16* __restrict__ wfrag,
                             int Cout, int Cin, int ntiles) {
  int idx = blockIdx.x * blockDim.x + threadIdx.x;
  int total = 9 * ntiles * 512;
  if (idx >= total) return;
  int i    = idx & 15;
  int lane = (idx >> 4) & 31;
  int rest = idx >> 9;
  int nt   = rest % ntiles;
  int tap  = rest / ntiles;
  int n = nt * 16 + (lane & 15);
  int k = (i < 8) ? ((lane >> 4) * 8 + i) : (16 + (lane >> 4) * 8 + (i - 8));  // lane<16 -> K 0..7,16..23; lane>=16 -> K 8..15,24..31
  float v = 0.f;
  if (n < Cout && k < Cin) v = w[((size_t)n * Cin + k) * 9 + tap];
  wfrag[idx] = (h16)v;
}

// ---------------------------------------------------------------------------
// Implicit-GEMM 3x3 conv via V_WMMA_F32_16X16X32_F16, branchless inner loop.
// Input is padded HWC f16 (zero halo), so all fragment loads are unconditional.
// One wave computes 64 pixels x 16 out-channels. K = 9 taps x 32 channels.
// act: 0 none, 1 relu, 2 sigmoid, 3 tanh.  out = act(conv*sc + bi)
// ---------------------------------------------------------------------------
__global__ __launch_bounds__(32) void k_conv3x3_wmma(
    const h16* __restrict__ xhwc, const h16* __restrict__ wfrag,
    float* __restrict__ out, int H, int W, int Hout, int Wout,
    int stride, int Cout, const float* __restrict__ sc,
    const float* __restrict__ bi, int act) {
  const int lane   = threadIdx.x;
  const int mlocal = lane & 15;
  const int khalf  = lane >> 4;            // 0 or 1
  const int ntiles = gridDim.y;
  const int ntile  = blockIdx.y;
  const int P      = Hout * Wout;
  const int base   = blockIdx.x * 64;
  const int Wp     = W + 2;

  v8f acc[4];
  for (int s = 0; s < 4; ++s)
    for (int r = 0; r < 8; ++r) acc[s][r] = 0.f;

  // Per-subtile: padded-space base offset of tap (kh=0,kw=0); computed once.
  int bofs[4];
  int pvm[4];
  for (int s = 0; s < 4; ++s) {
    int p = base + s * 16 + mlocal;
    pvm[s] = (p < P);
    int pp = pvm[s] ? p : (P - 1);         // clamp: loads stay in-bounds, store masked
    int ph = pp / Wout, pw = pp % Wout;
    bofs[s] = (ph * stride * Wp + pw * stride) * CPAD + khalf * 8;
  }

  for (int kh = 0; kh < 3; ++kh) {
    for (int kw = 0; kw < 3; ++kw) {
      const int tap = kh * 3 + kw;
      const int tofs = (kh * Wp + kw) * CPAD;       // wave-uniform
      const h16* wp = wfrag + ((size_t)(tap * ntiles + ntile) * 32 + lane) * 16;
      v16h bf = *(const v16h*)wp;
      if (tap < 8)
        __builtin_prefetch((const void*)(wfrag + ((size_t)((tap + 1) * ntiles + ntile) * 32 + lane) * 16), 0, 0);
      for (int s = 0; s < 4; ++s) {
        const h16* xp = xhwc + (size_t)(bofs[s] + tofs);
        v8h lo = *(const v8h*)xp;          // K = khalf*8 + 0..7
        v8h hi = *(const v8h*)(xp + 16);   // K = 16 + khalf*8 + 0..7
        v16h af = __builtin_shufflevector(lo, hi, 0, 1, 2, 3, 4, 5, 6, 7,
                                          8, 9, 10, 11, 12, 13, 14, 15);
        acc[s] = __builtin_amdgcn_wmma_f32_16x16x32_f16(false, af, false, bf,
                                                        (short)0, acc[s], false, false);
      }
    }
  }

  const int N = ntile * 16 + mlocal;
  if (N < Cout) {
    const float s_ = sc ? sc[N] : 1.f;
    const float b_ = bi ? bi[N] : 0.f;
    for (int s = 0; s < 4; ++s)
      for (int r = 0; r < 8; ++r) {
        int M = khalf * 8 + r;
        int p = base + s * 16 + M;
        if (p < P) {
          float v = acc[s][r] * s_ + b_;
          if (act == 1) v = fmaxf(v, 0.f);
          else if (act == 2) v = 1.f / (1.f + expf(-v));
          else if (act == 3) v = tanhf(v);
          out[(size_t)N * P + p] = v;
        }
      }
  }
}

// Transposed conv 3x3, stride 2, pad 1, out_pad 1 (scalar; small channel counts)
__global__ void k_convT(const float* __restrict__ x, const float* __restrict__ w,
                        float* __restrict__ out, int Cin, int Cout, int H, int W,
                        const float* __restrict__ sc, const float* __restrict__ bi,
                        const float* __restrict__ res, int resBefore, int doRelu) {
  int OH = 2 * H, OW = 2 * W;
  int idx = blockIdx.x * blockDim.x + threadIdx.x;
  if (idx >= Cout * OH * OW) return;
  int ow = idx % OW; int t = idx / OW; int oh = t % OH; int co = t / OH;
  float acc = 0.f;
  for (int kh = 0; kh < 3; ++kh) {
    int phh = oh + kh - 1;
    if (phh < 0 || (phh & 1) || (phh >> 1) >= H) continue;
    int ih = phh >> 1;
    for (int kw = 0; kw < 3; ++kw) {
      int pww = ow + kw - 1;
      if (pww < 0 || (pww & 1) || (pww >> 1) >= W) continue;
      int iw = pww >> 1;
      const float* xp = x + (size_t)ih * W + iw;
      const float* wp = w + (size_t)co * 9 + (2 - kh) * 3 + (2 - kw);
      float a = 0.f;
      for (int ci = 0; ci < Cin; ++ci)
        a += xp[(size_t)ci * H * W] * wp[(size_t)ci * Cout * 9];
      acc += a;
    }
  }
  float v = acc;
  if (sc) v *= sc[co];
  if (bi) v += bi[co];
  if (res && resBefore) v += res[idx];
  if (doRelu) v = fmaxf(v, 0.f);
  if (res && !resBefore) v = res[idx] + v;
  out[idx] = v;
}

// Softmax statistics over D: conf = max prob = 1/S, res = sum(prob*depth)
__global__ void k_softmax_stats(const float* __restrict__ logits, const float* __restrict__ dvals,
                                float* __restrict__ conf_out, float* __restrict__ res_out,
                                int D, int HW) {
  int hw = blockIdx.x * blockDim.x + threadIdx.x;
  if (hw >= HW) return;
  float M = -3.4e38f;
  for (int d = 0; d < D; ++d) M = fmaxf(M, logits[(size_t)d * HW + hw]);
  float S = 0.f, R = 0.f;
  for (int d = 0; d < D; ++d) {
    float e = expf(logits[(size_t)d * HW + hw] - M);
    S += e;
    R += e * dvals[(size_t)d * HW + hw];
  }
  conf_out[hw] = 1.f / S;
  res_out[hw] = R / S;
}

// pair_conf_all = concat([pc, tile(pc, D)]) -> 66 x HW
__global__ void k_broadcast_pca(const float* __restrict__ pc, float* __restrict__ outp) {
  int idx = blockIdx.x * blockDim.x + threadIdx.x;
  if (idx >= 66 * HWn) return;
  int i = idx / HWn, hw = idx % HWn;
  int src = (i < 2) ? i : ((i - 2) & 1);
  outp[idx] = pc[(size_t)src * HWn + hw];
}

// sim slice for depth d: confidence-weighted warped*ref, computed on the fly
__global__ void k_sim_slice(const float* __restrict__ ref, const float* __restrict__ sA,
                            const float* __restrict__ sB, const float* __restrict__ dvals,
                            const float* __restrict__ rt, const float* __restrict__ pc,
                            float* __restrict__ sim, int d, int C, int H, int W) {
  int HW = H * W;
  int idx = blockIdx.x * blockDim.x + threadIdx.x;
  if (idx >= C * HW) return;
  int hw = idx % HW; int c = idx / HW;
  int w = hw % W, h = hw / W;
  float dep = dvals[(size_t)d * HW + hw];
  float conf[2] = { pc[hw], pc[HW + hw] };
  float den = 1e-5f + conf[0] + conf[1];
  float refv = ref[(size_t)c * HW + hw];
  const float* srcs[2] = { sA, sB };
  float num = 0.f;
  for (int pv = 0; pv < 2; ++pv) {
    const float* R = rt + pv * 12;
    float rx = R[0] * w + R[1] * h + R[2];
    float ry = R[3] * w + R[4] * h + R[5];
    float rz = R[6] * w + R[7] * h + R[8];
    float pz = rz * dep + R[11];
    float px = (rx * dep + R[9]) / pz;
    float py = (ry * dep + R[10]) / pz;
    num += bisample(srcs[pv] + (size_t)c * HW, H, W, px, py) * refv * conf[pv];
  }
  sim[idx] = num / den;
}

__global__ void k_sig_split(const float* __restrict__ gates, const float* __restrict__ h,
                            float* __restrict__ z, float* __restrict__ rh, int Cg, int HW) {
  int i = blockIdx.x * blockDim.x + threadIdx.x;
  if (i >= Cg * HW) return;
  z[i] = 1.f / (1.f + expf(-gates[i]));
  float r = 1.f / (1.f + expf(-gates[(size_t)Cg * HW + i]));
  rh[i] = r * h[i];
}

__global__ void k_gru_update(float* __restrict__ h, const float* __restrict__ z,
                             const float* __restrict__ cand, int n) {
  int i = blockIdx.x * blockDim.x + threadIdx.x;
  if (i >= n) return;
  float zz = z[i];
  h[i] = (1.f - zz) * h[i] + zz * cand[i];
}

// pr=exp(rc); running max / depth-weighted sum / exp-sum with on-the-fly bilinear depth upsample
__global__ void k_accum(const float* __restrict__ rc, const float* __restrict__ dslice,
                        float* __restrict__ esum, float* __restrict__ dimg,
                        float* __restrict__ mx, int H, int W) {
  int OW_ = 2 * W;
  int idx = blockIdx.x * blockDim.x + threadIdx.x;
  if (idx >= 4 * H * W) return;
  int ow = idx % OW_, oh = idx / OW_;
  float pr = expf(rc[idx]);
  float sy = (oh + 0.5f) * 0.5f - 0.5f;
  float sx = (ow + 0.5f) * 0.5f - 0.5f;
  float y0f = floorf(sy), x0f = floorf(sx);
  float wy = sy - y0f, wx = sx - x0f;
  int y0 = (int)y0f, x0 = (int)x0f;
  int y0c = min(max(y0, 0), H - 1), y1c = min(max(y0 + 1, 0), H - 1);
  int x0c = min(max(x0, 0), W - 1), x1c = min(max(x0 + 1, 0), W - 1);
  float v00 = dslice[y0c * W + x0c], v10 = dslice[y0c * W + x1c];
  float v01 = dslice[y1c * W + x0c], v11 = dslice[y1c * W + x1c];
  float dd = v00 * (1 - wy) * (1 - wx) + v10 * (1 - wy) * wx + v01 * wy * (1 - wx) + v11 * wy * wx;
  mx[idx] = fmaxf(mx[idx], pr);
  dimg[idx] += dd * pr;
  esum[idx] += pr;
}

__global__ void k_finalize(const float* __restrict__ dimg, const float* __restrict__ esum,
                           const float* __restrict__ mx, float* __restrict__ odepth,
                           float* __restrict__ opmap, int n) {
  int i = blockIdx.x * blockDim.x + threadIdx.x;
  if (i >= n) return;
  float den = esum[i] + 1e-10f;
  odepth[i] = dimg[i] / den;
  opmap[i]  = mx[i] / den;
}

// ---------------------------------------------------------------------------
extern "C" void kernel_launch(void* const* d_in, const int* in_sizes, int n_in,
                              void* d_out, int out_size, void* d_ws, size_t ws_size,
                              hipStream_t stream) {
  (void)in_sizes; (void)n_in; (void)out_size; (void)ws_size;
  const float* features = (const float*)d_in[0];
  const float* pm       = (const float*)d_in[1];
  const float* dvals    = (const float*)d_in[2];
  auto F = [&](int i) { return (const float*)d_in[i]; };
  float* outf = (float*)d_out;

  const float* ref  = features;
  const float* src1 = features + (size_t)Cn * HWn;
  const float* src2 = features + 2 * (size_t)Cn * HWn;

  // ---- workspace layout (256B aligned slabs) ----
  float* wsf = (float*)d_ws;
  size_t off = 0;
  auto alloc = [&](size_t nfl) { float* p = wsf + off; off += (nfl + 63) & ~(size_t)63; return p; };
  float* rt     = alloc(24);
  float* cost   = alloc((size_t)Dn * HWn);
  float* c0b    = alloc(32 * (size_t)HWn);
  float* ta     = alloc(32 * (size_t)HW2);
  float* c2b    = alloc(32 * (size_t)HW2);
  float* tb     = alloc(32 * (size_t)HW4);
  float* c4b    = alloc(32 * (size_t)HW4);
  float* tc     = alloc(32 * (size_t)HW8);
  float* hhb    = alloc(32 * (size_t)HW8);
  float* y4     = alloc(32 * (size_t)HW4);
  float* y2     = alloc(32 * (size_t)HW2);
  float* y0     = alloc(32 * (size_t)HWn);
  float* logits = alloc((size_t)Dn * HWn);
  float* pc     = alloc(2 * (size_t)HWn);
  float* simd_  = alloc(32 * (size_t)HWn);
  float* c1b    = alloc((size_t)BC * HWn);
  float* gates1 = alloc(2 * (size_t)BC * HWn);
  float* z1     = alloc((size_t)BC * HWn);
  float* rh1    = alloc((size_t)BC * HWn);
  float* cand1  = alloc((size_t)BC * HWn);
  float* s1     = alloc((size_t)BC * HWn);
  float* c2s    = alloc(2 * (size_t)BC * HW2);
  float* gates2 = alloc(4 * (size_t)BC * HW2);
  float* z2     = alloc(2 * (size_t)BC * HW2);
  float* rh2    = alloc(2 * (size_t)BC * HW2);
  float* cand2  = alloc(2 * (size_t)BC * HW2);
  float* s2     = alloc(2 * (size_t)BC * HW2);
  float* ub     = alloc((size_t)BC * HWn);
  float* rcb    = alloc(OHW);
  float* esum   = alloc(OHW);
  float* dimg   = alloc(OHW);
  float* mxb    = alloc(OHW);
  h16* hwc      = (h16*)alloc((size_t)CPAD * PADPIX / 2);
  h16* wfrags   = (h16*)alloc(14 * (size_t)WSLOT / 2);

  auto EL = [&](int n) { return dim3((n + 255) / 256); };

  // ---- projection setup + weight fragment packing ----
  k_proj_setup<<<dim3(1), dim3(1), 0, stream>>>(pm, rt);
  struct WL { int widx, Cin, Cout, ntiles; };
  const WL wls[14] = {
    {4,32,32,2},{7,32,32,2},{10,32,32,2},{13,32,32,2},{16,32,32,2},{19,32,32,2},{22,32,32,2},
    {34,32,32,2},                                       // prob
    {36,32,8,1},{38,16,16,1},{40,16,8,1},{42,8,16,1},{44,32,32,2},{46,32,16,1}
  };
  for (int i = 0; i < 14; ++i) {
    int total = 9 * wls[i].ntiles * 512;
    k_pack_wfrag<<<EL(total), dim3(256), 0, stream>>>(F(wls[i].widx), wfrags + (size_t)i * WSLOT,
                                                      wls[i].Cout, wls[i].Cin, wls[i].ntiles);
  }

  auto pack = [&](const float* a, int Ca, const float* b, int Cb, int H, int W) {
    int total = (H + 2) * (W + 2) * CPAD;
    k_pack_hwc<<<EL(total), dim3(256), 0, stream>>>(a, Ca, b, Cb, hwc, H, W);
  };
  auto conv = [&](int slot, float* outp, int H, int W, int Hout, int Wout, int stride,
                  int Cout, int ntiles, const float* sc, const float* bi, int act) {
    dim3 g((Hout * Wout + 63) / 64, ntiles);
    k_conv3x3_wmma<<<g, dim3(32), 0, stream>>>(hwc, wfrags + (size_t)slot * WSLOT, outp,
                                               H, W, Hout, Wout, stride, Cout, sc, bi, act);
  };
  auto convT = [&](const float* x, const float* w, float* outp, int Cin, int Cout,
                   int H, int W, const float* sc, const float* bi,
                   const float* res, int resBefore, int relu) {
    int total = Cout * 4 * H * W;
    k_convT<<<EL(total), dim3(256), 0, stream>>>(x, w, outp, Cin, Cout, H, W, sc, bi, res, resBefore, relu);
  };

  // ---- per-pair: warp+cost, cost_reg U-Net, softmax stats ----
  const float* srcs[2] = { src1, src2 };
  for (int pv = 0; pv < 2; ++pv) {
    k_warp_cost<<<EL(Dn * HWn), dim3(256), 0, stream>>>(ref, srcs[pv], dvals, rt + pv * 12,
                                                        cost, Cn, Dn, Hn, Wn);
    pack(cost, 32, nullptr, 0, Hn, Wn);  conv(0, c0b, Hn, Wn, Hn, Wn, 1, 32, 2, F(5),  F(6),  1);
    pack(c0b, 32, nullptr, 0, Hn, Wn);   conv(1, ta,  Hn, Wn, H2, W2, 2, 32, 2, F(8),  F(9),  1);
    pack(ta,  32, nullptr, 0, H2, W2);   conv(2, c2b, H2, W2, H2, W2, 1, 32, 2, F(11), F(12), 1);
    pack(c2b, 32, nullptr, 0, H2, W2);   conv(3, tb,  H2, W2, H4, W4, 2, 32, 2, F(14), F(15), 1);
    pack(tb,  32, nullptr, 0, H4, W4);   conv(4, c4b, H4, W4, H4, W4, 1, 32, 2, F(17), F(18), 1);
    pack(c4b, 32, nullptr, 0, H4, W4);   conv(5, tc,  H4, W4, H8, W8, 2, 32, 2, F(20), F(21), 1);
    pack(tc,  32, nullptr, 0, H8, W8);   conv(6, hhb, H8, W8, H8, W8, 1, 32, 2, F(23), F(24), 1);
    convT(hhb, F(25), y4, 32, 32, H8, W8, F(26), F(27), c4b, 0, 1);   // c4 + relu(y*s+b)
    convT(y4,  F(28), y2, 32, 32, H4, W4, F(29), F(30), c2b, 0, 1);   // c2 + ...
    convT(y2,  F(31), y0, 32, 32, H2, W2, F(32), F(33), c0b, 0, 1);   // c0 + ...
    pack(y0, 32, nullptr, 0, Hn, Wn);    conv(7, logits, Hn, Wn, Hn, Wn, 1, 32, 2, nullptr, F(35), 0);
    k_softmax_stats<<<EL(HWn), dim3(256), 0, stream>>>(logits, dvals, pc + (size_t)pv * HWn,
                                                       outf + O_PRES + (size_t)pv * HWn, Dn, HWn);
  }
  k_broadcast_pca<<<EL(66 * HWn), dim3(256), 0, stream>>>(pc, outf + O_PCA);

  // ---- scan over depth slices: GRU fusion + soft-argmax accumulation ----
  k_zero<<<EL(BC * HWn),     dim3(256), 0, stream>>>(s1,   0.f, BC * HWn);
  k_zero<<<EL(2 * BC * HW2), dim3(256), 0, stream>>>(s2,   0.f, 2 * BC * HW2);
  k_zero<<<EL(OHW),          dim3(256), 0, stream>>>(esum, 0.f, OHW);
  k_zero<<<EL(OHW),          dim3(256), 0, stream>>>(dimg, 0.f, OHW);
  k_zero<<<EL(OHW),          dim3(256), 0, stream>>>(mxb,  0.f, OHW);

  for (int d = 0; d < Dn; ++d) {
    k_sim_slice<<<EL(Cn * HWn), dim3(256), 0, stream>>>(ref, src1, src2, dvals, rt, pc,
                                                        simd_, d, Cn, Hn, Wn);
    pack(simd_, 32, nullptr, 0, Hn, Wn); conv(8,  c1b,    Hn, Wn, Hn, Wn, 1, BC,     1, nullptr, F(37), 1);
    pack(c1b, BC, s1, BC, Hn, Wn);       conv(9,  gates1, Hn, Wn, Hn, Wn, 1, 2 * BC, 1, nullptr, F(39), 0);
    k_sig_split<<<EL(BC * HWn), dim3(256), 0, stream>>>(gates1, s1, z1, rh1, BC, HWn);
    pack(c1b, BC, rh1, BC, Hn, Wn);      conv(10, cand1,  Hn, Wn, Hn, Wn, 1, BC,     1, nullptr, F(41), 3);
    k_gru_update<<<EL(BC * HWn), dim3(256), 0, stream>>>(s1, z1, cand1, BC * HWn);
    pack(s1, BC, nullptr, 0, Hn, Wn);    conv(11, c2s,    Hn, Wn, H2, W2, 2, 2 * BC, 1, nullptr, F(43), 1);
    pack(c2s, 2 * BC, s2, 2 * BC, H2, W2); conv(12, gates2, H2, W2, H2, W2, 1, 4 * BC, 2, nullptr, F(45), 0);
    k_sig_split<<<EL(2 * BC * HW2), dim3(256), 0, stream>>>(gates2, s2, z2, rh2, 2 * BC, HW2);
    pack(c2s, 2 * BC, rh2, 2 * BC, H2, W2); conv(13, cand2, H2, W2, H2, W2, 1, 2 * BC, 1, nullptr, F(47), 3);
    k_gru_update<<<EL(2 * BC * HW2), dim3(256), 0, stream>>>(s2, z2, cand2, 2 * BC * HW2);
    convT(s2, F(48), ub,  2 * BC, BC, H2, W2, nullptr, F(49), s1, 1, 1); // relu(y+b+r1)
    convT(ub, F(50), rcb, BC, 1, Hn, Wn, nullptr, F(51), nullptr, 0, 0); // y+b
    k_accum<<<EL(OHW), dim3(256), 0, stream>>>(rcb, dvals + (size_t)d * HWn, esum, dimg, mxb, Hn, Wn);
  }
  k_finalize<<<EL(OHW), dim3(256), 0, stream>>>(dimg, esum, mxb, outf + O_DEPTH, outf + O_PMAP, OHW);
}
